// FractalApproximator_82463372083644
// MI455X (gfx1250) — matrix operand order repaired and checked by reading, not compiled, not run on hardware
//
#include <hip/hip_runtime.h>

typedef __attribute__((ext_vector_type(16))) _Float16 v16h;
typedef __attribute__((ext_vector_type(8)))  _Float16 v8h;
typedef __attribute__((ext_vector_type(2)))  _Float16 h2;
typedef __attribute__((ext_vector_type(8)))  float    v8f;

#define HID    64
#define ENC_K  32
#define DEC_N  10
#define WAVES  8
#define HPAD   72   // padded row stride (halves) for the h staging tile

// Native V_TANH_F32 on CDNA5 (probe-confirmed last round); rcp fallback kept for safety.
#if __has_builtin(__builtin_amdgcn_tanhf)
__device__ __forceinline__ float fast_tanh(float x) { return __builtin_amdgcn_tanhf(x); }
#elif __has_builtin(__builtin_amdgcn_tanh_f32)
__device__ __forceinline__ float fast_tanh(float x) { return __builtin_amdgcn_tanh_f32(x); }
#else
__device__ __forceinline__ float fast_tanh(float x) {
    float e = __expf(2.0f * x);
    return 1.0f - 2.0f * __builtin_amdgcn_rcpf(e + 1.0f);
}
#endif

#if __has_builtin(__builtin_amdgcn_fractf)
__device__ __forceinline__ float fract_f(float v) { return __builtin_amdgcn_fractf(v); }
#else
__device__ __forceinline__ float fract_f(float v) { return v - floorf(v); }
#endif

// Packed-f16 SELU on a pair of f32 accumulators (cvt_pk -> pk_max/min -> v_exp_f16 -> pk_fma).
__device__ __forceinline__ h2 selu_h2(float lo, float hi) {
    h2 y; y.x = (_Float16)lo; y.y = (_Float16)hi;
    const h2 zero  = {(_Float16)0.0f,          (_Float16)0.0f};
    const h2 lg2e  = {(_Float16)1.442695041f,  (_Float16)1.442695041f};
    const h2 l2    = {(_Float16)1.05070099f,   (_Float16)1.05070099f};   // lambda
    const h2 la2   = {(_Float16)1.75809934f,   (_Float16)1.75809934f};   // lambda*alpha
    h2 p = __builtin_elementwise_max(y, zero);
    h2 q = __builtin_elementwise_min(y, zero);
    h2 e = __builtin_elementwise_exp2(q * lg2e);
    return __builtin_elementwise_fma(l2, p, __builtin_elementwise_fma(la2, e, -la2));
}

// Butterfly add over the 16 lanes of each half-wave via ds_swizzle (xor patterns).
__device__ __forceinline__ float xsum16(float v) {
    v += __int_as_float(__builtin_amdgcn_ds_swizzle(__float_as_int(v), 0x041F)); // xor 1
    v += __int_as_float(__builtin_amdgcn_ds_swizzle(__float_as_int(v), 0x081F)); // xor 2
    v += __int_as_float(__builtin_amdgcn_ds_swizzle(__float_as_int(v), 0x101F)); // xor 4
    v += __int_as_float(__builtin_amdgcn_ds_swizzle(__float_as_int(v), 0x201F)); // xor 8
    return v;
}

__global__ __launch_bounds__(256) void fractal_mlp_kernel(
    const float* __restrict__ x,
    const float* __restrict__ We, const float* __restrict__ be,
    const float* __restrict__ W1, const float* __restrict__ b1,
    const float* __restrict__ W2, const float* __restrict__ b2,
    const float* __restrict__ Wd, const float* __restrict__ bd,
    float* __restrict__ out, int B)
{
    __shared__ __attribute__((aligned(32))) _Float16 sWe[HID * ENC_K];
    __shared__ __attribute__((aligned(32))) _Float16 sW1[HID * HID];
    __shared__ __attribute__((aligned(32))) _Float16 sW2[HID * HID];
    __shared__ __attribute__((aligned(32))) _Float16 sWd[16 * HID];
    __shared__ float sbe2[HID], sb1[HID], sb2[HID], sbd[16];
    __shared__ __attribute__((aligned(32))) _Float16 sH[WAVES][16 * HPAD];

    const int tid = threadIdx.x;
    // ---- convert weights to f16 in LDS (B-matrix-friendly: row n, contiguous k) ----
    for (int i = tid; i < HID * ENC_K; i += 256) sWe[i] = (_Float16)We[i];
    for (int i = tid; i < HID * HID;  i += 256) sW1[i] = (_Float16)W1[i];
    for (int i = tid; i < HID * HID;  i += 256) sW2[i] = (_Float16)W2[i];
    for (int i = tid; i < 16 * HID;   i += 256) {
        int n = i / HID;
        sWd[i] = (n < DEC_N) ? (_Float16)Wd[i] : (_Float16)0.0f;
    }
    if (tid < HID) { sbe2[tid] = 2.0f * be[tid]; sb1[tid] = b1[tid]; sb2[tid] = b2[tid]; }
    if (tid < 16)  { sbd[tid] = (tid < DEC_N) ? bd[tid] : 0.0f; }
    __syncthreads();

    const int wave = tid >> 5;
    const int lane = tid & 31;
    const int g    = lane >> 4;   // half-wave group (K-split of A / B fragments)
    const int nl   = lane & 15;   // M index for A, N index for B/C/D
    _Float16* hbuf = &sH[wave][0];

    const float dscale = (nl < DEC_N) ? __powf(10.0f, 1.0f - (float)nl) : 0.0f;

    // ---- preload ALL loop-invariant operands into registers ----
    v16h bWe[4], bW1[4][2], bW2[4][2], bWd[2];
    float be2v[4], b1v[4], b2v[4];
    #pragma unroll
    for (int t = 0; t < 4; ++t) {
        const int n = t * 16 + nl;
        bWe[t] = *(const v16h*)&sWe[n * ENC_K + g * 16];
        #pragma unroll
        for (int c = 0; c < 2; ++c) {
            bW1[t][c] = *(const v16h*)&sW1[n * HID + c * 32 + g * 16];
            bW2[t][c] = *(const v16h*)&sW2[n * HID + c * 32 + g * 16];
        }
        be2v[t] = sbe2[n]; b1v[t] = sb1[n]; b2v[t] = sb2[n];
    }
    bWd[0] = *(const v16h*)&sWd[nl * HID + g * 16];
    bWd[1] = *(const v16h*)&sWd[nl * HID + 32 + g * 16];
    const float bdv = sbd[nl];

    // Load the 16x64 A-matrix (two K-chunk fragments) from the staging tile.
    auto load_a = [&](v16h& a0, v16h& a1) {
        const _Float16* row = hbuf + nl * HPAD;
        v8h l0 = *(const v8h*)(row + g * 16);
        v8h h0 = *(const v8h*)(row + 8 + g * 16);
        v8h l1 = *(const v8h*)(row + 32 + g * 16);
        v8h h1 = *(const v8h*)(row + 40 + g * 16);
        #pragma unroll
        for (int i = 0; i < 8; ++i) {
            a0[i] = l0[i]; a0[i + 8] = h0[i];
            a1[i] = l1[i]; a1[i + 8] = h1[i];
        }
    };
    // One hidden layer: 8 WMMAs + packed-f16 SELU, result back into the staging tile.
    auto hidden_layer = [&](const v16h (&bw)[4][2], const float (&bias)[4]) {
        v16h a0, a1;
        load_a(a0, a1);
        v8f acc[4];
        #pragma unroll
        for (int t = 0; t < 4; ++t) {
            v8f c = {};
            c = __builtin_amdgcn_wmma_f32_16x16x32_f16(
                false, a0, false, bw[t][0], (short)0, c, false, false);
            acc[t] = __builtin_amdgcn_wmma_f32_16x16x32_f16(
                false, a1, false, bw[t][1], (short)0, c, false, false);
        }
        #pragma unroll
        for (int t = 0; t < 4; ++t) {
            const int n = t * 16 + nl;
            #pragma unroll
            for (int i = 0; i < 4; ++i) {
                h2 y = selu_h2(acc[t][2*i] + bias[t], acc[t][2*i + 1] + bias[t]);
                hbuf[(2*i     + 8*g) * HPAD + n] = y.x;
                hbuf[(2*i + 1 + 8*g) * HPAD + n] = y.y;
            }
        }
        __asm__ volatile("s_wait_dscnt 0x0" ::: "memory");
    };

    const int ngroups = B >> 4;   // 16 samples per wave-group
    for (int grp = blockIdx.x * WAVES + wave; grp < ngroups; grp += gridDim.x * WAVES) {
        const int s0 = grp << 4;

        // ---- digit extraction: each lane runs BOTH features' chains for sample s0+nl ----
        const float2 xv = *(const float2*)(x + (size_t)(s0 + nl) * 2);
        float v0 = xv.x * 0.1f;
        float v1 = xv.y * 0.1f;
        float dsum[ENC_K];
        #pragma unroll
        for (int k = 0; k < ENC_K; ++k) {
            dsum[k] = v0 + v1;                    // shared encoder Linear: sum digit streams
            v0 = fract_f(v0) * 10.0f;
            v1 = fract_f(v1) * 10.0f;
        }

        // ---- encoder A fragment straight from registers (K = e + 16g) ----
        v16h aenc;
        #pragma unroll
        for (int e = 0; e < 8; ++e) {
            aenc[e]     = (_Float16)(g ? dsum[e + 16] : dsum[e]);
            aenc[e + 8] = (_Float16)(g ? dsum[e + 24] : dsum[e + 8]);
        }

        // ---- encoder: h = tanh(D @ We^T + 2*be), K=32 -> one WMMA per N-tile ----
        {
            v8f acc[4];
            #pragma unroll
            for (int t = 0; t < 4; ++t) {
                v8f c = {};
                acc[t] = __builtin_amdgcn_wmma_f32_16x16x32_f16(
                    false, aenc, false, bWe[t], (short)0, c, false, false);
            }
            #pragma unroll
            for (int t = 0; t < 4; ++t) {
                const int n = t * 16 + nl;
                #pragma unroll
                for (int r = 0; r < 8; ++r) {
                    const int m = r + 8 * g;
                    hbuf[m * HPAD + n] = (_Float16)fast_tanh(acc[t][r] + be2v[t]);
                }
            }
        }
        __asm__ volatile("s_wait_dscnt 0x0" ::: "memory");

        // ---- hidden layers (unrolled; register-resident weights) ----
        hidden_layer(bW1, b1v);
        hidden_layer(bW2, b2v);

        // ---- decoder: mags = tanh(h @ Wd^T + bd); val = mags @ 10^(1-n) ----
        {
            v16h a0, a1;
            load_a(a0, a1);
            v8f c = {};
            c = __builtin_amdgcn_wmma_f32_16x16x32_f16(
                false, a0, false, bWd[0], (short)0, c, false, false);
            c = __builtin_amdgcn_wmma_f32_16x16x32_f16(
                false, a1, false, bWd[1], (short)0, c, false, false);
            #pragma unroll
            for (int r = 0; r < 8; ++r) {
                float contrib = xsum16(fast_tanh(c[r] + bdv) * dscale);
                if (nl == r) {
                    const int m = s0 + r + 8 * g;
                    float2 o; o.x = contrib; o.y = contrib;  // both output features identical
                    *(float2*)&out[(size_t)m * 2] = o;
                }
            }
        }
    }
}

extern "C" void kernel_launch(void* const* d_in, const int* in_sizes, int n_in,
                              void* d_out, int out_size, void* d_ws, size_t ws_size,
                              hipStream_t stream) {
    const float* x  = (const float*)d_in[0];
    const float* We = (const float*)d_in[1];
    const float* be = (const float*)d_in[2];
    const float* W1 = (const float*)d_in[3];
    const float* b1 = (const float*)d_in[4];
    const float* W2 = (const float*)d_in[5];
    const float* b2 = (const float*)d_in[6];
    const float* Wd = (const float*)d_in[7];
    const float* bd = (const float*)d_in[8];
    float* out = (float*)d_out;
    (void)d_ws; (void)ws_size; (void)out_size; (void)n_in;

    const int B = in_sizes[0] / 2;
    const int ngroups = B / 16;
    int blocks = (ngroups + WAVES - 1) / WAVES;
    if (blocks > 1024) blocks = 1024;
    fractal_mlp_kernel<<<dim3(blocks), dim3(256), 0, stream>>>(
        x, We, be, W1, b1, W2, b2, Wd, bd, out, B);
}
